// SE3EquivariantAttention_55396488184161
// MI455X (gfx1250) — matrix operand order repaired and checked by reading, not compiled, not run on hardware
//
#include <hip/hip_runtime.h>
#include <hip/hip_bf16.h>
#include <math.h>

typedef __attribute__((ext_vector_type(16))) _Float16 v16h;
typedef __attribute__((ext_vector_type(8)))  float    v8f;
typedef __attribute__((ext_vector_type(4)))  unsigned int v4u;
typedef __attribute__((ext_vector_type(8)))  int      v8i;
typedef __attribute__((ext_vector_type(4)))  int      v4i;

#define DEVINL __device__ __forceinline__

constexpr int Bc = 2;
constexpr int Nc = 384;
constexpr int EDGE_Hc = 64;
constexpr int Hc = 8;
constexpr long long NEDGE = (long long)Bc * Nc * Nc;   // 294912

DEVINL v8f wmma_f16(v16h a, v16h b, v8f c) {
  return __builtin_amdgcn_wmma_f32_16x16x32_f16(
      /*neg_a=*/false, a, /*neg_b=*/false, b,
      /*c_mod=*/(short)0, c, /*reuse_a=*/false, /*reuse_b=*/false);
}

// A fragment: 16x32 f16, ISA 7.12.2 layout.
// lane L: M = L&15 ; K = (v<4?0:16) + (L>>4)*8 + (v&3)*2 + s
template <typename FA>
DEVINL v16h load_afrag(int lane, FA f) {
  v16h a;
  int m = lane & 15, kh = (lane >> 4) * 8;
#pragma unroll
  for (int v = 0; v < 8; ++v) {
    int kb = ((v & 4) << 2) + kh + (v & 3) * 2;
    a[2 * v]     = f(m, kb);
    a[2 * v + 1] = f(m, kb + 1);
  }
  return a;
}

// B fragment: 32x16 f16. lane L: N = L&15 ; K = (v<4?0:16) + (L>>4)*8 + (v&3)*2 + s
template <typename FB>
DEVINL v16h load_bfrag(int lane, FB f) {
  v16h b;
  int n = lane & 15, kh = (lane >> 4) * 8;
#pragma unroll
  for (int v = 0; v < 8; ++v) {
    int kb = ((v & 4) << 2) + kh + (v & 3) * 2;
    b[2 * v]     = f(kb, n);
    b[2 * v + 1] = f(kb + 1, n);
  }
  return b;
}

DEVINL float lrelu(float v) { return v > 0.f ? v : 0.01f * v; }

// ---- Tensor Data Mover: DMA a 2-D f32 tile (tile_h rows x tile_w elems) into LDS.
// D# packing per CDNA5 ISA ch.8 (group0: count/lds/global/type, group1: dims+strides).
// This toolchain exposes the 6-arg builtin: (g0 u32x4, g1 i32x8, g2 i32x4, g3 i32x4, i32x8, cpol)
DEVINL void tdm_load_2d_f32(const void* gptr, unsigned lds_addr,
                            unsigned tile_w, unsigned tile_h,
                            unsigned tensor_w, unsigned tensor_h,
                            unsigned row_stride) {
  unsigned long long ga = (unsigned long long)(size_t)gptr;
  v4u g0;
  g0[0] = 1u;                                              // count=1 (valid user D#)
  g0[1] = lds_addr;                                        // lds_addr[31:0]
  g0[2] = (unsigned)(ga & 0xFFFFFFFFu);                    // global_addr[31:0]
  g0[3] = (unsigned)((ga >> 32) & 0x1FFFFFFu) | (2u << 30); // global_addr[56:32] | type=2
  v8i g1;
  g1[0] = (int)(2u << 16);                                 // data_size=4B, wg_mask=0
  g1[1] = (int)((tensor_w & 0xFFFFu) << 16);               // tensor_dim0[15:0]
  g1[2] = (int)(((tensor_w >> 16) & 0xFFFFu) | ((tensor_h & 0xFFFFu) << 16));
  g1[3] = (int)(((tensor_h >> 16) & 0xFFFFu) | ((tile_w & 0xFFFFu) << 16)); // tile_dim0
  g1[4] = (int)(tile_h & 0xFFFFu);                         // tile_dim1 (tile_dim2=0)
  g1[5] = (int)row_stride;                                 // tensor_dim0_stride[31:0]
  g1[6] = 0;                                               // stride0 hi / stride1 lo
  g1[7] = 0;
  v4i z4 = {0, 0, 0, 0};
  v8i z8 = {0, 0, 0, 0, 0, 0, 0, 0};
  __builtin_amdgcn_tensor_load_to_lds(g0, g1, z4, z4, z8, 0);
}

// ---------------------------------------------------------------- K0: weights->f16
__global__ void k_cvt_weights(const float* Wqk, const float* We, const float* Wn,
                              const float* Wed, const float* Wc1, const float* Wc2,
                              _Float16* wq, _Float16* we, _Float16* wn,
                              _Float16* wed, _Float16* wc1, _Float16* wc2) {
  int i = blockIdx.x * blockDim.x + threadIdx.x;
  if (i < 128 * 512) wq[i]  = (_Float16)Wqk[i];
  if (i < 128 * 256) we[i]  = (_Float16)We[i];
  if (i < 256 * 128) wn[i]  = (_Float16)Wn[i];
  if (i < 128 * 64)  wed[i] = (_Float16)Wed[i];
  if (i < 128 * 128) wc1[i] = (_Float16)Wc1[i];
  if (i < 128)       wc2[i] = (_Float16)Wc2[i];
}

// ---------------------------------------------------------------- K1: qk = x@W_qk + b
__global__ void k_qk(const float* __restrict__ x, const _Float16* __restrict__ wq,
                     const float* __restrict__ bqk, float* __restrict__ qk) {
  int wid  = blockIdx.x * (blockDim.x >> 5) + (threadIdx.x >> 5);  // 48*32 tiles
  int lane = threadIdx.x & 31;
  int m0 = (wid >> 5) * 16, n0 = (wid & 31) * 16;
  v8f c = {};
#pragma unroll
  for (int k0 = 0; k0 < 128; k0 += 32) {
    v16h a = load_afrag(lane, [&](int m, int k) { return (_Float16)x[(m0 + m) * 128 + k0 + k]; });
    v16h b = load_bfrag(lane, [&](int k, int n) { return wq[(k0 + k) * 512 + n0 + n]; });
    c = wmma_f16(a, b, c);
  }
  int m = m0 + (lane >> 4) * 8, n = n0 + (lane & 15);
#pragma unroll
  for (int r = 0; r < 8; ++r) qk[(m + r) * 512 + n] = c[r] + bqk[n];
}

// ---------------------------------------------------------------- K2: edge MLP (big GEMM)
__global__ void k_edge_mlp(const float* __restrict__ edge_attr, const float* __restrict__ coor,
                           const _Float16* __restrict__ we, const float* __restrict__ be,
                           _Float16* __restrict__ e_m, _Float16* __restrict__ e_v) {
  __shared__ float    EA[16][64];     // TDM-staged edge_attr tile
  __shared__ _Float16 As[16][128];    // f16 A tile = [edge_attr | rbf]
  long long row0 = (long long)blockIdx.x * 16;
  int b  = (int)(row0 / ((long long)Nc * Nc));
  int i  = (int)((row0 / Nc) % Nc);
  int j0 = (int)(row0 % Nc);
  int tid = threadIdx.x;

  // Wave 0 issues the Tensor Data Mover DMA for the 16x64 f32 edge_attr tile.
  if (tid < 32) {
    tdm_load_2d_f32(edge_attr + row0 * EDGE_Hc,
                    (unsigned)(size_t)(void*)&EA[0][0],
                    /*tile_w=*/64, /*tile_h=*/16,
                    /*tensor_w=*/64, /*tensor_h=*/16,
                    /*row_stride=*/64);
    __builtin_amdgcn_s_wait_tensorcnt(0);
  }
  __syncthreads();

  {  // stage A = [edge_attr | rbf(dist)] in LDS (f16)
    int m = tid >> 3, c0 = (tid & 7) << 4;
    int j = j0 + m;
    const float* ci = coor + ((long long)b * Nc + i) * 3;
    const float* cj = coor + ((long long)b * Nc + j) * 3;
    float dx = ci[0] - cj[0], dy = ci[1] - cj[1], dz = ci[2] - cj[2];
    float dist = fminf(sqrtf(dx * dx + dy * dy + dz * dz), 2.0f);
    const float step  = 2.0f / 63.0f;
    const float coeff = -0.5f / (step * step);
#pragma unroll
    for (int c = 0; c < 16; ++c) {
      int k = c0 + c;
      float v;
      if (k < 64) v = EA[m][k];
      else { float d = dist - (float)(k - 64) * step; v = __expf(coeff * d * d); }
      As[m][k] = (_Float16)v;
    }
  }
  __syncthreads();

  int lane = tid & 31, w = tid >> 5;
  v16h a[4];
#pragma unroll
  for (int ks = 0; ks < 4; ++ks)
    a[ks] = load_afrag(lane, [&](int m, int k) { return As[m][ks * 32 + k]; });
  long long ebase = row0 * 128;
#pragma unroll
  for (int t4 = 0; t4 < 4; ++t4) {
    int n0 = (w + 4 * t4) * 16;
    v8f c = {};
#pragma unroll
    for (int ks = 0; ks < 4; ++ks) {
      v16h bb = load_bfrag(lane, [&](int k, int n) { return we[(ks * 32 + k) * 256 + n0 + n]; });
      c = wmma_f16(a[ks], bb, c);
    }
    int mrow = (lane >> 4) * 8, ng = n0 + (lane & 15);
    _Float16* dst = (ng < 128) ? e_m : e_v;
    int nn = ng & 127;
    float bias = be[ng];
#pragma unroll
    for (int r = 0; r < 8; ++r)
      dst[ebase + (long long)(mrow + r) * 128 + nn] = (_Float16)lrelu(c[r] + bias);
  }
}

// ---------------------------------------------------------------- K3: trilinear logits
__global__ void k_logits(const float* __restrict__ qk, const _Float16* __restrict__ e_m,
                         const int* __restrict__ emask, float* __restrict__ att) {
  long long idx = (long long)blockIdx.x * blockDim.x + threadIdx.x;
  if (idx >= NEDGE * Hc) return;
  int h = (int)(idx & 7);
  long long e = idx >> 3;
  int j = (int)(e % Nc);
  int i = (int)((e / Nc) % Nc);
  int b = (int)(e / ((long long)Nc * Nc));
  const float* lm = qk + ((long long)b * Nc + i) * 512 + h * 16;
  const float* rm = qk + ((long long)b * Nc + j) * 512 + 128 + h * 16;
  const _Float16* em = e_m + e * 128 + h * 16;
  float s = 0.f;
#pragma unroll
  for (int d = 0; d < 16; ++d) s += lm[d] * rm[d] * (float)em[d];
  s *= 0.25f;  // 1/sqrt(D)
  if (emask[e] == 0) s = -3.402823466e38f;
  att[idx] = s;
}

// ---------------------------------------------------------------- K4: softmax over j
__global__ void k_softmax(float* __restrict__ att) {
  int wid  = blockIdx.x * (blockDim.x >> 5) + (threadIdx.x >> 5);  // (b,i,h) rows: 6144
  int lane = threadIdx.x & 31;
  int h = wid & 7;
  int bi = wid >> 3;
  float* base = att + ((long long)bi * Nc) * 8 + h;
  float vals[12];
  float mx = -3.402823466e38f;
#pragma unroll
  for (int c = 0; c < 12; ++c) { vals[c] = base[(lane + 32 * c) * 8]; mx = fmaxf(mx, vals[c]); }
#pragma unroll
  for (int o = 16; o; o >>= 1) mx = fmaxf(mx, __shfl_xor(mx, o, 32));
  float sum = 0.f;
#pragma unroll
  for (int c = 0; c < 12; ++c) { vals[c] = __expf(vals[c] - mx); sum += vals[c]; }
#pragma unroll
  for (int o = 16; o; o >>= 1) sum += __shfl_xor(sum, o, 32);
  float inv = 1.0f / sum;
#pragma unroll
  for (int c = 0; c < 12; ++c) base[(lane + 32 * c) * 8] = vals[c] * inv;
}

// ---------------------------------------------------------------- K5: coor_att gate
__global__ void k_coor_att(const float* __restrict__ qk, const _Float16* __restrict__ e_v,
                           const float* __restrict__ att, const _Float16* __restrict__ wc1,
                           const float* __restrict__ bc1, const _Float16* __restrict__ wc2,
                           const float* __restrict__ bc2, const int* __restrict__ emask,
                           float* __restrict__ catt) {
  __shared__ _Float16 AV[16][128];
  __shared__ float Hd[16][128];
  long long row0 = (long long)blockIdx.x * 16;
  int b  = (int)(row0 / ((long long)Nc * Nc));
  int i  = (int)((row0 / Nc) % Nc);
  int j0 = (int)(row0 % Nc);
  int tid = threadIdx.x;
  {  // stage av = soft * l_v * r_v * e_v
    int m = tid >> 3, c0 = (tid & 7) << 4;
    long long erow = row0 + m;
    const float* lv = qk + ((long long)b * Nc + i) * 512 + 256;
    const float* rv = qk + ((long long)b * Nc + j0 + m) * 512 + 384;
    const _Float16* ev = e_v + erow * 128;
    const float* sm = att + erow * 8;
#pragma unroll
    for (int c = 0; c < 16; ++c) {
      int k = c0 + c;
      AV[m][k] = (_Float16)(sm[k >> 4] * lv[k] * rv[k] * (float)ev[k]);
    }
  }
  __syncthreads();
  int lane = tid & 31, w = tid >> 5;
  v16h a[4];
#pragma unroll
  for (int ks = 0; ks < 4; ++ks)
    a[ks] = load_afrag(lane, [&](int m, int k) { return AV[m][ks * 32 + k]; });
#pragma unroll
  for (int nt2 = 0; nt2 < 2; ++nt2) {
    int n0 = (w * 2 + nt2) * 16;
    v8f c = {};
#pragma unroll
    for (int ks = 0; ks < 4; ++ks) {
      v16h bb = load_bfrag(lane, [&](int k, int n) { return wc1[(ks * 32 + k) * 128 + n0 + n]; });
      c = wmma_f16(a[ks], bb, c);
    }
    int mrow = (lane >> 4) * 8, n = n0 + (lane & 15);
    float bias = bc1[n];
#pragma unroll
    for (int r = 0; r < 8; ++r) Hd[mrow + r][n] = lrelu(c[r] + bias);
  }
  __syncthreads();
  if (tid < 16) {
    float acc = bc2[0];
    for (int k = 0; k < 128; ++k) acc += Hd[tid][k] * (float)wc2[k];
    long long erow = row0 + tid;
    catt[erow] = (emask[erow] != 0) ? acc : 0.0f;
  }
}

// ---------------------------------------------------------------- K6: edge_out
__global__ void k_edge_out(const _Float16* __restrict__ e_v, const float* __restrict__ att,
                           const _Float16* __restrict__ wed, const float* __restrict__ bed,
                           float* __restrict__ out_edge) {
  __shared__ _Float16 EV[16][128];
  long long row0 = (long long)blockIdx.x * 16;
  int tid = threadIdx.x;
  {
    int m = tid >> 3, c0 = (tid & 7) << 4;
    long long erow = row0 + m;
    const _Float16* ev = e_v + erow * 128;
    const float* sm = att + erow * 8;
#pragma unroll
    for (int c = 0; c < 16; ++c) {
      int k = c0 + c;
      EV[m][k] = (_Float16)((float)ev[k] * sm[k >> 4]);
    }
  }
  __syncthreads();
  int lane = tid & 31, w = tid >> 5;
  int n0 = w * 16;
  v8f c = {};
#pragma unroll
  for (int ks = 0; ks < 4; ++ks) {
    v16h a  = load_afrag(lane, [&](int m, int k) { return EV[m][ks * 32 + k]; });
    v16h bb = load_bfrag(lane, [&](int k, int n) { return wed[(ks * 32 + k) * 64 + n0 + n]; });
    c = wmma_f16(a, bb, c);
  }
  int mrow = (lane >> 4) * 8, n = n0 + (lane & 15);
  float bias = bed[n];
#pragma unroll
  for (int r = 0; r < 8; ++r)
    out_edge[(row0 + mrow + r) * 64 + n] = c[r] + bias;
}

// ---------------------------------------------------------------- K7: node einsum (softmax @ [l_v|r_v])
__global__ void k_node_gather(const float* __restrict__ att, const float* __restrict__ qk,
                              float* __restrict__ node_mid) {
  int wid  = blockIdx.x * (blockDim.x >> 5) + (threadIdx.x >> 5);  // 2*8*24*2 = 768 waves
  int lane = threadIdx.x & 31;
  int nt   = wid & 1;
  int rest = wid >> 1;
  int mt = rest % 24;
  int h  = (rest / 24) % 8;
  int b  = rest / 192;
  int m0 = mt * 16;
  long long abase = (long long)b * Nc * Nc * 8;
  int vbase = 256 + nt * 128 + h * 16;
  v8f c = {};
#pragma unroll 4
  for (int k0 = 0; k0 < Nc; k0 += 32) {
    v16h a = load_afrag(lane, [&](int m, int k) {
      return (_Float16)att[abase + ((long long)(m0 + m) * Nc + (k0 + k)) * 8 + h];
    });
    v16h bb = load_bfrag(lane, [&](int k, int n) {
      return (_Float16)qk[((long long)b * Nc + k0 + k) * 512 + vbase + n];
    });
    c = wmma_f16(a, bb, c);
  }
  int m = m0 + (lane >> 4) * 8, n = lane & 15;
#pragma unroll
  for (int r = 0; r < 8; ++r)
    node_mid[((long long)b * Nc + m + r) * 256 + h * 32 + nt * 16 + n] = c[r];
}

// ---------------------------------------------------------------- K8: node_out GEMM
__global__ void k_node_out(const float* __restrict__ node_mid, const _Float16* __restrict__ wn,
                           const float* __restrict__ bn, float* __restrict__ out_node) {
  int wid  = blockIdx.x * (blockDim.x >> 5) + (threadIdx.x >> 5);  // 48*8 tiles
  int lane = threadIdx.x & 31;
  int m0 = (wid >> 3) * 16, n0 = (wid & 7) * 16;
  v8f c = {};
#pragma unroll
  for (int k0 = 0; k0 < 256; k0 += 32) {
    v16h a  = load_afrag(lane, [&](int m, int k) { return (_Float16)node_mid[(m0 + m) * 256 + k0 + k]; });
    v16h bb = load_bfrag(lane, [&](int k, int n) { return wn[(k0 + k) * 128 + n0 + n]; });
    c = wmma_f16(a, bb, c);
  }
  int m = m0 + (lane >> 4) * 8, n = n0 + (lane & 15);
#pragma unroll
  for (int r = 0; r < 8; ++r)
    out_node[(m + r) * 128 + n] = c[r] + bn[n];
}

// ---------------------------------------------------------------- K9: coordinate aggregation
__global__ void k_coor_out(const float* __restrict__ catt, const float* __restrict__ coor,
                           const int* __restrict__ emask, const int* __restrict__ flex,
                           float* __restrict__ out_coor) {
  int wid  = blockIdx.x * (blockDim.x >> 5) + (threadIdx.x >> 5);  // b*384+i : 768 waves
  int lane = threadIdx.x & 31;
  int b = wid / Nc;
  const float* ci = coor + (long long)wid * 3;
  long long rbase = (long long)wid * Nc;
  float ax = 0.f, ay = 0.f, az = 0.f, ms = 0.f;
  for (int j = lane; j < Nc; j += 32) {
    float ca = catt[rbase + j];
    const float* cj = coor + ((long long)b * Nc + j) * 3;
    float dx = ci[0] - cj[0], dy = ci[1] - cj[1], dz = ci[2] - cj[2];
    float nrm = sqrtf(dx * dx + dy * dy + dz * dz);
    float den = (nrm == 0.f) ? 1e8f : fmaxf(nrm, 1e-8f);
    float inv = ca / den;
    ax += dx * inv; ay += dy * inv; az += dz * inv;
    ms += (float)emask[rbase + j];
  }
#pragma unroll
  for (int o = 16; o; o >>= 1) {
    ax += __shfl_xor(ax, o, 32); ay += __shfl_xor(ay, o, 32);
    az += __shfl_xor(az, o, 32); ms += __shfl_xor(ms, o, 32);
  }
  if (lane == 0) {
    float s = (float)flex[wid] / (ms + 1e-7f);
    out_coor[wid * 3 + 0] = ax * s;
    out_coor[wid * 3 + 1] = ay * s;
    out_coor[wid * 3 + 2] = az * s;
  }
}

extern "C" void kernel_launch(void* const* d_in, const int* in_sizes, int n_in,
                              void* d_out, int out_size, void* d_ws, size_t ws_size,
                              hipStream_t stream) {
  (void)in_sizes; (void)n_in; (void)out_size; (void)ws_size;
  const float* x        = (const float*)d_in[0];
  const float* coor     = (const float*)d_in[1];
  const float* edge_attr= (const float*)d_in[2];
  const float* W_qk     = (const float*)d_in[3];
  const float* b_qk     = (const float*)d_in[4];
  const float* W_e      = (const float*)d_in[5];
  const float* b_e      = (const float*)d_in[6];
  const float* W_node   = (const float*)d_in[7];
  const float* b_node   = (const float*)d_in[8];
  const float* W_edge   = (const float*)d_in[9];
  const float* b_edge   = (const float*)d_in[10];
  const float* W_c1     = (const float*)d_in[11];
  const float* b_c1     = (const float*)d_in[12];
  const float* W_c2     = (const float*)d_in[13];
  const float* b_c2     = (const float*)d_in[14];
  const int*   edge_mask= (const int*)d_in[15];
  const int*   flex_mask= (const int*)d_in[16];

  float* out_node = (float*)d_out;                       // B*N*128
  float* out_edge = out_node + (long long)Bc * Nc * 128; // B*N*N*64
  float* out_coor = out_edge + NEDGE * 64;               // B*N*3

  char* w = (char*)d_ws;
  float*    qk      = (float*)w;     w += (long long)Bc * Nc * 512 * 4;
  _Float16* e_m     = (_Float16*)w;  w += NEDGE * 128 * 2;
  _Float16* e_v     = (_Float16*)w;  w += NEDGE * 128 * 2;
  float*    att     = (float*)w;     w += NEDGE * 8 * 4;
  float*    catt    = (float*)w;     w += NEDGE * 4;
  float*    node_mid= (float*)w;     w += (long long)Bc * Nc * 256 * 4;
  _Float16* wq_h    = (_Float16*)w;  w += 128 * 512 * 2;
  _Float16* we_h    = (_Float16*)w;  w += 128 * 256 * 2;
  _Float16* wn_h    = (_Float16*)w;  w += 256 * 128 * 2;
  _Float16* wed_h   = (_Float16*)w;  w += 128 * 64 * 2;
  _Float16* wc1_h   = (_Float16*)w;  w += 128 * 128 * 2;
  _Float16* wc2_h   = (_Float16*)w;  w += 128 * 2;

  const long long ntile = NEDGE / 16;  // 18432 edge tiles

  k_cvt_weights<<<256, 256, 0, stream>>>(W_qk, W_e, W_node, W_edge, W_c1, W_c2,
                                         wq_h, we_h, wn_h, wed_h, wc1_h, wc2_h);
  k_qk<<<384, 128, 0, stream>>>(x, wq_h, b_qk, qk);
  k_edge_mlp<<<(int)ntile, 128, 0, stream>>>(edge_attr, coor, we_h, b_e, e_m, e_v);
  k_logits<<<(int)((NEDGE * 8) / 256), 256, 0, stream>>>(qk, e_m, edge_mask, att);
  k_softmax<<<1536, 128, 0, stream>>>(att);
  k_coor_att<<<(int)ntile, 128, 0, stream>>>(qk, e_v, att, wc1_h, b_c1, wc2_h, b_c2,
                                             edge_mask, catt);
  k_edge_out<<<(int)ntile, 128, 0, stream>>>(e_v, att, wed_h, b_edge, out_edge);
  k_node_gather<<<192, 128, 0, stream>>>(att, qk, node_mid);
  k_node_out<<<96, 128, 0, stream>>>(node_mid, wn_h, b_node, out_node);
  k_coor_out<<<192, 128, 0, stream>>>(catt, coor, edge_mask, flex_mask, out_coor);
}
